// GraphAttentionLayer_13005160972779
// MI455X (gfx1250) — matrix operand order repaired and checked by reading, not compiled, not run on hardware
//
#include <hip/hip_runtime.h>

// GAT layer for MI455X (gfx1250, wave32, WMMA).
// Identity: attention logits are constant per row after masking, so
//   out = LayerNorm( (Adj @ (x@W)) / rowsum(Adj) ) * gamma + beta.
// GEMM2 (Adj @ H, 8.6 GFLOP, 64 MB adjacency traffic) dominates; its K-loop
// LDS path is a flat 16 KB tile copy (async global->LDS when available)
// because H/W are pre-swizzled into the WMMA B-fragment memory order.

typedef __attribute__((ext_vector_type(16))) _Float16 v16h;
typedef __attribute__((ext_vector_type(8)))  _Float16 v8h;
typedef __attribute__((ext_vector_type(8)))  float    v8f;
typedef int v4i __attribute__((vector_size(16)));   // matches builtin param type

#define NNODES 4096
#define IN_F   256
#define OUT_F  256
#define LN_EPS 1e-5f

// ---- optional CDNA5 async global->LDS path (fallback: manual copy) --------
#if defined(__gfx1250__) && \
    __has_builtin(__builtin_amdgcn_global_load_async_to_lds_b128) && \
    __has_builtin(__builtin_amdgcn_s_wait_asynccnt)
#define ASYNC_TILES 1
#define GLB_V4I(p) ((__attribute__((address_space(1))) v4i*)(p))
#define LDS_V4I(p) ((__attribute__((address_space(3))) v4i*)(p))
#else
#define ASYNC_TILES 0
#endif

// Copy one 16 KB tile (8192 f16) from global to LDS; 64 B per thread (t<256).
__device__ __forceinline__ void tile_copy_issue(const _Float16* __restrict__ gsrc,
                                                _Float16* lds, int t) {
#if ASYNC_TILES
  const _Float16* g = gsrc + t * 32;
  _Float16* l = lds + t * 32;
  __builtin_amdgcn_global_load_async_to_lds_b128(GLB_V4I(g),      LDS_V4I(l),      0, 0);
  __builtin_amdgcn_global_load_async_to_lds_b128(GLB_V4I(g + 8),  LDS_V4I(l + 8),  0, 0);
  __builtin_amdgcn_global_load_async_to_lds_b128(GLB_V4I(g + 16), LDS_V4I(l + 16), 0, 0);
  __builtin_amdgcn_global_load_async_to_lds_b128(GLB_V4I(g + 24), LDS_V4I(l + 24), 0, 0);
#else
  const v8h* src = (const v8h*)gsrc + t * 4;
  v8h* dst = (v8h*)lds + t * 4;
  v8h r0 = src[0], r1 = src[1], r2 = src[2], r3 = src[3];
  dst[0] = r0; dst[1] = r1; dst[2] = r2; dst[3] = r3;
#endif
}

__device__ __forceinline__ void tile_copy_fence() {
#if ASYNC_TILES
  __builtin_amdgcn_s_wait_asynccnt(0);
#endif
}

// Swizzled B-fragment memory order for a [K][256] f16 matrix:
//   flat = (krow>>5)*8192 + (col>>4)*512 + (col&15)*32 + (krow&31)
// so each 32-row K-tile is one contiguous 16 KB block whose layout inside LDS
// is directly addressable as [ncg][n16][k] for v16h fragment reads.

// ---------------------------------------------------------------------------
// Kernel 0: pack W (256x256 f32) into swizzled f16 Wsw. One-time, tiny.
// ---------------------------------------------------------------------------
__global__ __launch_bounds__(256)
void gat_pack_w(const float* __restrict__ W, _Float16* __restrict__ Wsw) {
  const int idx  = blockIdx.x * 256 + threadIdx.x;   // 0..65535
  const int krow = idx >> 8;
  const int col  = idx & 255;
  const int dst  = ((krow >> 5) << 13) + ((col >> 4) << 9) + ((col & 15) << 5) + (krow & 31);
  Wsw[dst] = (_Float16)W[idx];
}

// ---------------------------------------------------------------------------
// Kernel 1: Hsw = swizzled f16(x @ W). 64 blocks x 256 threads, 64x256 tile.
// ---------------------------------------------------------------------------
__global__ __launch_bounds__(256)
void gat_gemm_xw(const float* __restrict__ x, const _Float16* __restrict__ Wsw,
                 _Float16* __restrict__ Hsw) {
  __shared__ __align__(64) _Float16 ldsA[64][32];   // x tile row-major
  __shared__ __align__(64) _Float16 ldsB[8192];     // W tile, fragment order

  const int t    = threadIdx.x;
  const int lane = t & 31, wave = t >> 5;
  const int msub = wave & 3, nhalf = wave >> 2;
  const int g    = lane >> 4, n16 = lane & 15;
  const int mBase = blockIdx.x * 64;

  v8f acc[8];
  #pragma unroll
  for (int i = 0; i < 8; ++i) { v8f z = {}; acc[i] = z; }

  for (int kb = 0; kb < IN_F / 32; ++kb) {
    __syncthreads();
    tile_copy_issue(Wsw + (size_t)kb * 8192, ldsB, t);
    { // x tile 64x32 f32 -> f16 LDS (single 16B store per thread).
      const int row = t >> 2, c = t & 3;
      const float* src = x + (size_t)(mBase + row) * IN_F + kb * 32 + c * 8;
      const float4 f0 = ((const float4*)src)[0];
      const float4 f1 = ((const float4*)src)[1];
      v8h xv;
      xv[0] = (_Float16)f0.x; xv[1] = (_Float16)f0.y;
      xv[2] = (_Float16)f0.z; xv[3] = (_Float16)f0.w;
      xv[4] = (_Float16)f1.x; xv[5] = (_Float16)f1.y;
      xv[6] = (_Float16)f1.z; xv[7] = (_Float16)f1.w;
      *(v8h*)&ldsA[row][c * 8] = xv;
    }
    tile_copy_fence();
    __syncthreads();

    const int mloc = msub * 16 + n16;
    const v8h a0 = *(const v8h*)&ldsA[mloc][16 * g];
    const v8h a1 = *(const v8h*)&ldsA[mloc][16 * g + 8];
    const v16h afrag = __builtin_shufflevector(a0, a1,
        0,1,2,3,4,5,6,7,8,9,10,11,12,13,14,15);
    #pragma unroll
    for (int nt = 0; nt < 8; ++nt) {
      const int ncg = nhalf * 8 + nt;
      const v16h bfrag = *(const v16h*)&ldsB[(ncg * 16 + n16) * 32 + 16 * g];
      acc[nt] = __builtin_amdgcn_wmma_f32_16x16x32_f16(
          false, afrag, false, bfrag, (short)0, acc[nt], false, false);
    }
  }

  // Store Hsw: accumulator rows v=0..7 are contiguous k-locs -> one 16B store.
  const int rowBase = msub * 16 + g * 8;            // 0..56, within 64-row tile
  const int kbOut   = (mBase + rowBase) >> 5;
  const int kloc    = (mBase + rowBase) & 31;
  #pragma unroll
  for (int nt = 0; nt < 8; ++nt) {
    const int col = (nhalf * 8 + nt) * 16 + n16;
    v8h h;
    #pragma unroll
    for (int v = 0; v < 8; ++v) h[v] = (_Float16)acc[nt][v];
    *(v8h*)&Hsw[(size_t)kbOut * 8192 + ((col >> 4) << 9) + ((col & 15) << 5) + kloc] = h;
  }
}

// ---------------------------------------------------------------------------
// Kernel 2: out = LN( (Adj @ H) / deg ) * gamma + beta.  Adjacency read once.
// ---------------------------------------------------------------------------
__global__ __launch_bounds__(256)
void gat_attn_ln(const int* __restrict__ adj, const _Float16* __restrict__ Hsw,
                 const float* __restrict__ gamma, const float* __restrict__ beta,
                 float* __restrict__ out) {
  __shared__ __align__(64) _Float16 ldsA[64][32];   // adj tile as f16 0/1
  __shared__ __align__(64) _Float16 ldsB[8192];     // H tile, fragment order
  __shared__ float degLds[256];
  __shared__ float rowS1[64][2];
  __shared__ float rowS2[64][2];
  __shared__ float rowMu[64], rowRstd[64], rowInvDeg[64];

  const int t    = threadIdx.x;
  const int lane = t & 31, wave = t >> 5;
  const int msub = wave & 3, nhalf = wave >> 2;
  const int g    = lane >> 4, n16 = lane & 15;
  const int mBase = blockIdx.x * 64;

  v8f acc[8];
  #pragma unroll
  for (int i = 0; i < 8; ++i) { v8f z = {}; acc[i] = z; }
  int degAcc = 0;

  for (int kb = 0; kb < NNODES / 32; ++kb) {
    __syncthreads();
    tile_copy_issue(Hsw + (size_t)kb * 8192, ldsB, t);
    { // adj tile 64x32 int32 -> packed f16 0/1, fused degree partial sum.
      const int row = t >> 2, c = t & 3;
      const int* src = adj + (size_t)(mBase + row) * NNODES + kb * 32 + c * 8;
      const int4 i0 = ((const int4*)src)[0];
      const int4 i1 = ((const int4*)src)[1];
      const int b0 = i0.x > 0, b1 = i0.y > 0, b2 = i0.z > 0, b3 = i0.w > 0;
      const int b4 = i1.x > 0, b5 = i1.y > 0, b6 = i1.z > 0, b7 = i1.w > 0;
      degAcc += b0 + b1 + b2 + b3 + b4 + b5 + b6 + b7;
      uint4 p;   // two f16 (1.0h = 0x3C00) per dword, built with selects
      p.x = (b0 ? 0x3C00u : 0u) | (b1 ? 0x3C000000u : 0u);
      p.y = (b2 ? 0x3C00u : 0u) | (b3 ? 0x3C000000u : 0u);
      p.z = (b4 ? 0x3C00u : 0u) | (b5 ? 0x3C000000u : 0u);
      p.w = (b6 ? 0x3C00u : 0u) | (b7 ? 0x3C000000u : 0u);
      *(uint4*)&ldsA[row][c * 8] = p;
    }
    tile_copy_fence();
    __syncthreads();

    const int mloc = msub * 16 + n16;
    const v8h a0 = *(const v8h*)&ldsA[mloc][16 * g];
    const v8h a1 = *(const v8h*)&ldsA[mloc][16 * g + 8];
    const v16h afrag = __builtin_shufflevector(a0, a1,
        0,1,2,3,4,5,6,7,8,9,10,11,12,13,14,15);
    #pragma unroll
    for (int nt = 0; nt < 8; ++nt) {
      const int ncg = nhalf * 8 + nt;
      const v16h bfrag = *(const v16h*)&ldsB[(ncg * 16 + n16) * 32 + 16 * g];
      acc[nt] = __builtin_amdgcn_wmma_f32_16x16x32_f16(
          false, afrag, false, bfrag, (short)0, acc[nt], false, false);
    }
  }

  // ---- Fused epilogue: degree + LayerNorm statistics + store ----
  degLds[t] = (float)degAcc;

  float s1[8], s2[8];
  #pragma unroll
  for (int v = 0; v < 8; ++v) {
    float a = 0.f, b = 0.f;
    #pragma unroll
    for (int nt = 0; nt < 8; ++nt) { const float c = acc[nt][v]; a += c; b += c * c; }
    s1[v] = a; s2[v] = b;
  }
  // Reduce across the 16 lanes of each half (wave32 xor shuffles, masks < 16
  // never mix the two M-halves).
  #pragma unroll
  for (int m = 1; m <= 8; m <<= 1) {
    #pragma unroll
    for (int v = 0; v < 8; ++v) {
      s1[v] += __shfl_xor(s1[v], m, 32);
      s2[v] += __shfl_xor(s2[v], m, 32);
    }
  }
  if (n16 == 0) {                       // lanes 0 and 16 publish per-row stats
    #pragma unroll
    for (int v = 0; v < 8; ++v) {
      const int rloc = msub * 16 + g * 8 + v;
      rowS1[rloc][nhalf] = s1[v];
      rowS2[rloc][nhalf] = s2[v];
    }
  }
  __syncthreads();

  if (t < 64) {
    const float d = degLds[4 * t] + degLds[4 * t + 1] + degLds[4 * t + 2] + degLds[4 * t + 3];
    const float inv = d > 0.f ? 1.f / d : 0.f;
    const float s1t = rowS1[t][0] + rowS1[t][1];
    const float s2t = rowS2[t][0] + rowS2[t][1];
    const float mu  = s1t * inv * (1.f / OUT_F);
    const float ex2 = s2t * inv * inv * (1.f / OUT_F);
    rowMu[t]     = mu;
    rowRstd[t]   = rsqrtf(ex2 - mu * mu + LN_EPS);
    rowInvDeg[t] = inv;
  }
  __syncthreads();

  #pragma unroll
  for (int nt = 0; nt < 8; ++nt) {
    const int col = (nhalf * 8 + nt) * 16 + n16;
    const float gm = gamma[col], bt = beta[col];
    #pragma unroll
    for (int v = 0; v < 8; ++v) {
      const int rloc = msub * 16 + g * 8 + v;
      const float val = acc[nt][v] * rowInvDeg[rloc];
      out[(size_t)(mBase + rloc) * OUT_F + col] =
          (val - rowMu[rloc]) * rowRstd[rloc] * gm + bt;
    }
  }
}

// ---------------------------------------------------------------------------
extern "C" void kernel_launch(void* const* d_in, const int* in_sizes, int n_in,
                              void* d_out, int out_size, void* d_ws, size_t ws_size,
                              hipStream_t stream) {
  const float* x     = (const float*)d_in[0];
  const int*   adj   = (const int*)d_in[1];
  const float* W     = (const float*)d_in[2];
  // d_in[3] ('a') is mathematically irrelevant: attention logits are constant
  // per row, so softmax reduces to adj[i,j]/deg(i) regardless of 'a'.
  const float* gamma = (const float*)d_in[4];
  const float* beta  = (const float*)d_in[5];

  _Float16* Hsw = (_Float16*)d_ws;                       // 2 MB, fully rewritten
  _Float16* Wsw = (_Float16*)d_ws + (size_t)NNODES * OUT_F;  // +128 KB

  gat_pack_w <<<IN_F * OUT_F / 256, 256, 0, stream>>>(W, Wsw);
  gat_gemm_xw<<<NNODES / 64, 256, 0, stream>>>(x, Wsw, Hsw);
  gat_attn_ln<<<NNODES / 64, 256, 0, stream>>>(adj, Hsw, gamma, beta, (float*)d_out);
}
